// AgentModel_30717606101642
// MI455X (gfx1250) — matrix operand order repaired and checked by reading, not compiled, not run on hardware
//
#include <hip/hip_runtime.h>
#include <hip/hip_bf16.h>
#include <math.h>

// ---------------- problem constants ----------------
#define B_N   65536
#define N_ITER 41
#define N_GRID 100
#define ROWS  64          // rows per block
#define TPB   512         // 16 waves of 32
#define XS    264         // padded LDS row stride (halves): 256 + 8

typedef __bf16 bf16_t;
typedef __bf16 v16bf __attribute__((ext_vector_type(16)));
typedef __bf16 v8bf  __attribute__((ext_vector_type(8)));
typedef float  v8f   __attribute__((ext_vector_type(8)));

__device__ inline float frcp(float x) { return __builtin_amdgcn_rcpf(x); }

// ---------------- weight prep: fp32 [K][N] -> bf16 [N][K] ----------------
__global__ void prep_weights_kernel(const float* __restrict__ pW1,
                                    const float* __restrict__ pW2,
                                    const float* __restrict__ vW1,
                                    const float* __restrict__ vW2,
                                    bf16_t* __restrict__ Wt) {
  int idx = blockIdx.x * blockDim.x + threadIdx.x;   // 0 .. 4*65536-1
  int mat = idx >> 16;
  int r   = idx & 65535;
  int n   = r >> 8;
  int k   = r & 255;
  const float* src = (mat == 0) ? pW1 : (mat == 1) ? pW2 : (mat == 2) ? vW1 : vW2;
  Wt[mat * 65536 + n * 256 + k] = (bf16_t)src[k * 256 + n];
}

// ---------------- async global->LDS copy of one 256x256 bf16 matrix ----------------
__device__ inline void async_issue_w(const bf16_t* __restrict__ gsrc,
                                     bf16_t* __restrict__ ldst, int t) {
  for (int i = t; i < 8192; i += TPB) {       // 8192 x 16B chunks
    int n = i >> 5;                           // row 0..255
    int c = i & 31;                           // 16B chunk in row
    unsigned lds = (unsigned)(uintptr_t)(ldst + n * XS + c * 8);
    unsigned off = (unsigned)(n * 512 + c * 16);
    asm volatile("global_load_async_to_lds_b128 %0, %1, %2"
                 :: "v"(lds), "v"(off), "s"(gsrc) : "memory");
  }
}
__device__ inline void async_wait() {
  asm volatile("s_wait_asynccnt 0" ::: "memory");
}

// ---------------- one 256-wide MLP layer for this block's 64-row tile ----------------
// 16 waves; wave w owns row-tile (w>>2) and nt = (w&3)*4 .. +3.
__device__ inline void mlp_layer(const bf16_t* __restrict__ Abuf,
                                 const bf16_t* __restrict__ Wbuf,
                                 bf16_t* __restrict__ Obuf,
                                 const float* __restrict__ bias,
                                 int w, int l) {
  const int m  = l & 15;     // M (and N) index within tile
  const int hf = l >> 4;     // lane half
  const int rt = w >> 2;     // row tile 0..3
  const int nt0 = (w & 3) * 4;
  const int arow = rt * 16 + m;
  for (int nt = nt0; nt < nt0 + 4; ++nt) {
    v8f acc = {0.f, 0.f, 0.f, 0.f, 0.f, 0.f, 0.f, 0.f};
    const float bv = bias[nt * 16 + m];
    #pragma unroll
    for (int kt = 0; kt < 8; ++kt) {
      // A fragment: lane holds row M=l%16; halves 0..7 -> K=kt*32+hf*8..,
      //             halves 8..15 -> K=kt*32+16+hf*8..  (ISA 7.12.2)
      const int ka = kt * 32 + hf * 8;
      v8bf alo = *(const v8bf*)(Abuf + arow * XS + ka);
      v8bf ahi = *(const v8bf*)(Abuf + arow * XS + ka + 16);
      // B fragment: lane holds N=l%16; same K order as A (kt*32+hf*8.., kt*32+16+hf*8..)
      const int kb = kt * 32 + hf * 8;
      v8bf blo = *(const v8bf*)(Wbuf + (nt * 16 + m) * XS + kb);
      v8bf bhi = *(const v8bf*)(Wbuf + (nt * 16 + m) * XS + kb + 16);
      v16bf a, b;
      #pragma unroll
      for (int i = 0; i < 8; ++i) {
        a[i] = alo[i]; a[i + 8] = ahi[i];
        b[i] = blo[i]; b[i + 8] = bhi[i];
      }
      acc = __builtin_amdgcn_wmma_f32_16x16x32_bf16(
          false, a, false, b, (short)0, acc, false, false);
    }
    // D layout: lane l, reg r -> M=(l/16)*8+r, N=l%16
    #pragma unroll
    for (int r = 0; r < 8; ++r) {
      float v = acc[r] + bv;
      v = v > 0.f ? v : 0.f;                   // relu (layers 1 & 2)
      Obuf[(rt * 16 + hf * 8 + r) * XS + nt * 16 + m] = (bf16_t)v;
    }
  }
}

// ---------------- scalar math helpers ----------------
__device__ inline float softplus_c(float x) {
  x = fmaxf(x, -5.f);
  return (x > 20.f) ? x : log1pf(expf(x));
}
__device__ inline float safe_pow(float z, float e) {
  return (z > 0.f) ? __expf(e * __logf(z)) : 0.f;
}

// ---------------- fused kernel ----------------
__global__ __launch_bounds__(TPB, 1)
void agent_fused_kernel(const float* __restrict__ x,
                        const bf16_t* __restrict__ Wt,
                        const float* __restrict__ pb1, const float* __restrict__ pb2,
                        const float* __restrict__ pW3, const float* __restrict__ pb3,
                        const float* __restrict__ vb1, const float* __restrict__ vb2,
                        const float* __restrict__ vW3, const float* __restrict__ vb3,
                        float* __restrict__ out) {
  __shared__ __align__(16) bf16_t s_x[ROWS * XS];
  __shared__ __align__(16) unsigned char s_wraw[256 * XS * 2];  // weights, later cdf/pdf/coeffs
  __shared__ __align__(16) bf16_t s_h1[ROWS * XS];
  __shared__ __align__(16) bf16_t s_h2[ROWS * XS];
  __shared__ float s_theta[ROWS * 5];
  __shared__ float s_rowc[ROWS * 8];
  __shared__ float s_inv[ROWS];

  bf16_t* s_w  = (bf16_t*)s_wraw;
  float*  s_cdf = (float*)s_wraw;                              // [ROWS][N_GRID]      (25600 B)
  float*  s_pdf = (float*)(s_wraw + ROWS * N_GRID * 4);        // [ROWS][101]         (25856 B)
  float2* s_cf  = (float2*)(s_wraw + 61440);                   // [ROWS][2][N_ITER]   (41984 B)

  const int t = threadIdx.x;
  const int w = t >> 5, l = t & 31;
  const long rowbase = (long)blockIdx.x * ROWS;

  // ---- overlap: start async W1p copy, then stage x tile (fp32 -> bf16) ----
  async_issue_w(Wt + 0 * 65536, s_w, t);
  for (int i = t; i < ROWS * 64; i += TPB) {                  // 64 float4 per row
    int row = i >> 6, c4 = i & 63;
    const float4 v = ((const float4*)(x + (rowbase + row) * 256))[c4];
    bf16_t* d = s_x + row * XS + c4 * 4;
    d[0] = (bf16_t)v.x; d[1] = (bf16_t)v.y; d[2] = (bf16_t)v.z; d[3] = (bf16_t)v.w;
  }
  async_wait();
  __syncthreads();

  // ---- policy MLP ----
  mlp_layer(s_x,  s_w, s_h1, pb1, w, l); __syncthreads();
  async_issue_w(Wt + 1 * 65536, s_w, t); async_wait(); __syncthreads();
  mlp_layer(s_h1, s_w, s_h2, pb2, w, l); __syncthreads();

  // ---- overlap: start async W1v copy, compute theta = h2 @ pW3 + pb3 ----
  async_issue_w(Wt + 2 * 65536, s_w, t);
  for (int i = t; i < ROWS * 5; i += TPB) {
    int row = i / 5, j = i % 5;
    float s = pb3[j];
    for (int k = 0; k < 256; ++k) s += (float)s_h2[row * XS + k] * pW3[k * 5 + j];
    s_theta[i] = s;
  }
  async_wait();
  __syncthreads();

  // ---- value MLP ----
  mlp_layer(s_x,  s_w, s_h1, vb1, w, l); __syncthreads();
  async_issue_w(Wt + 3 * 65536, s_w, t); async_wait(); __syncthreads();
  mlp_layer(s_h1, s_w, s_h2, vb2, w, l); __syncthreads();

  // ---- v output + per-row beta constants ----
  if (t < ROWS) {
    float s = vb3[0];
    for (int k = 0; k < 256; ++k) s += (float)s_h2[t * XS + k] * vW3[k];
    out[(long)B_N * 101 + rowbase + t] = frcp(1.f + expf(-s));

    float t0 = s_theta[t * 5 + 0], t1 = s_theta[t * 5 + 1], t2 = s_theta[t * 5 + 2];
    float a = softplus_c(s_theta[t * 5 + 3]);
    float b = softplus_c(s_theta[t * 5 + 4]);
    float mx = fmaxf(t0, fmaxf(t1, t2));
    float e0 = expf(t0 - mx), e1 = expf(t1 - mx), e2 = expf(t2 - mx);
    float inv = frcp(e0 + e1 + e2);
    float beta = expf(lgammaf(a) + lgammaf(b) - lgammaf(a + b));
    s_rowc[t * 8 + 0] = a;
    s_rowc[t * 8 + 1] = b;
    s_rowc[t * 8 + 2] = (a + 1.f) / (a + b + 2.f);   // S0
    s_rowc[t * 8 + 3] = 1.f / (a * beta);            // low-side scale
    s_rowc[t * 8 + 4] = 1.f / (b * beta);            // high-side scale
    s_rowc[t * 8 + 5] = e0 * inv;
    s_rowc[t * 8 + 6] = e1 * inv;
    s_rowc[t * 8 + 7] = e2 * inv;
  }
  __syncthreads();   // weights dead: s_wraw now cdf/pdf/coeff scratch

  // ---- precompute continued-fraction coefficients per (row, side, k) ----
  for (int i = t; i < ROWS * 2 * N_ITER; i += TPB) {
    int row  = i / (2 * N_ITER);
    int rem  = i % (2 * N_ITER);
    int side = rem / N_ITER;
    int k    = rem % N_ITER;
    float a = s_rowc[row * 8 + 0], b = s_rowc[row * 8 + 1];
    float p = side ? b : a, q = side ? a : b;
    float kk = (float)k;
    float c1 = -(p + kk) * (p + q + kk) / ((p + 2.f * kk) * (p + 2.f * kk + 1.f));
    float c2 = (k > 0) ? kk * (q - kk) / ((p + 2.f * kk - 1.f) * (p + 2.f * kk)) : 0.f;
    s_cf[(row * 2 + side) * N_ITER + k] = make_float2(c1, c2);
  }
  __syncthreads();

  // ---- beta cdf: branchless dual-chain continued fraction (ILP=2) ----
  const int NTASK = ROWS * N_GRID;
  for (int i = t; i < NTASK; i += 2 * TPB) {
    const int  iA   = i;
    const bool hasB = (i + TPB) < NTASK;
    const int  iB   = hasB ? (i + TPB) : iA;

    int rowA = iA / N_GRID, gA = iA % N_GRID;
    int rowB = iB / N_GRID, gB = iB % N_GRID;
    float xgA = (float)gA * (1.0f / 99.0f);
    float xgB = (float)gB * (1.0f / 99.0f);
    float S0A = s_rowc[rowA * 8 + 2], S0B = s_rowc[rowB * 8 + 2];
    int   sdA = (xgA > S0A), sdB = (xgB > S0B);
    float zA  = sdA ? 1.f - xgA : xgA;
    float zB  = sdB ? 1.f - xgB : xgB;
    const float2* cfA = s_cf + (rowA * 2 + sdA) * N_ITER;
    const float2* cfB = s_cf + (rowB * 2 + sdB) * N_ITER;

    float TA = 0.f, TB = 0.f;
    for (int k = N_ITER - 1; k > 0; --k) {
      float2 cA = cfA[k], cB = cfB[k];
      TA = cA.x * zA * frcp(TA + 1.f);
      TB = cB.x * zB * frcp(TB + 1.f);
      TA = cA.y * zA * frcp(TA + 1.f);
      TB = cB.y * zB * frcp(TB + 1.f);
    }
    TA = cfA[0].x * zA * frcp(TA + 1.f);
    TB = cfB[0].x * zB * frcp(TB + 1.f);

    float aA = s_rowc[rowA * 8 + 0], bA = s_rowc[rowA * 8 + 1];
    float PA = safe_pow(xgA, aA) * safe_pow(1.f - xgA, bA)
             * s_rowc[rowA * 8 + 3 + sdA] * frcp(TA + 1.f);
    float cdfA = sdA ? 1.f - PA : PA;
    s_cdf[iA] = fminf(fmaxf(cdfA, 0.f), 1.f);

    if (hasB) {
      float aB = s_rowc[rowB * 8 + 0], bB = s_rowc[rowB * 8 + 1];
      float PB = safe_pow(xgB, aB) * safe_pow(1.f - xgB, bB)
               * s_rowc[rowB * 8 + 3 + sdB] * frcp(TB + 1.f);
      float cdfB = sdB ? 1.f - PB : PB;
      s_cdf[iB] = fminf(fmaxf(cdfB, 0.f), 1.f);
    }
  }
  __syncthreads();

  // ---- assemble pdf ----
  for (int i = t; i < ROWS * 101; i += TPB) {
    int row = i / 101, c = i % 101;
    float p;
    if (c == 0)        p = s_rowc[row * 8 + 5];
    else if (c == 100) p = s_rowc[row * 8 + 6];
    else p = (s_cdf[row * N_GRID + c] - s_cdf[row * N_GRID + c - 1]) * s_rowc[row * 8 + 7];
    s_pdf[i] = p;
  }
  __syncthreads();

  if (t < ROWS) {
    float s = 0.f;
    for (int j = 0; j < 101; ++j) s += s_pdf[t * 101 + j];
    s_inv[t] = frcp(s);
  }
  __syncthreads();

  for (int i = t; i < ROWS * 101; i += TPB) {
    int row = i / 101;
    out[(rowbase + row) * 101 + (i % 101)] = s_pdf[i] * s_inv[row];
  }
}

// ---------------- launch ----------------
extern "C" void kernel_launch(void* const* d_in, const int* in_sizes, int n_in,
                              void* d_out, int out_size, void* d_ws, size_t ws_size,
                              hipStream_t stream) {
  (void)in_sizes; (void)n_in; (void)out_size; (void)ws_size;
  const float* x   = (const float*)d_in[0];
  const float* pW1 = (const float*)d_in[1];
  const float* pb1 = (const float*)d_in[2];
  const float* pW2 = (const float*)d_in[3];
  const float* pb2 = (const float*)d_in[4];
  const float* pW3 = (const float*)d_in[5];
  const float* pb3 = (const float*)d_in[6];
  const float* vW1 = (const float*)d_in[7];
  const float* vb1 = (const float*)d_in[8];
  const float* vW2 = (const float*)d_in[9];
  const float* vb2 = (const float*)d_in[10];
  const float* vW3 = (const float*)d_in[11];
  const float* vb3 = (const float*)d_in[12];

  bf16_t* Wt = (bf16_t*)d_ws;   // 4 * 256*256 bf16 = 512 KB

  prep_weights_kernel<<<(4 * 65536) / 256, 256, 0, stream>>>(pW1, pW2, vW1, vW2, Wt);
  agent_fused_kernel<<<B_N / ROWS, TPB, 0, stream>>>(
      x, Wt, pb1, pb2, pW3, pb3, vb1, vb2, vW3, vb3, (float*)d_out);
}
